// TemporalConvSNNEncoder_49134425866791
// MI455X (gfx1250) — matrix operand order repaired and checked by reading, not compiled, not run on hardware
//
#include <hip/hip_runtime.h>
#include <hip/hip_bf16.h>

typedef __attribute__((ext_vector_type(16))) _Float16 v16h;
typedef __attribute__((ext_vector_type(8)))  _Float16 v8h;
typedef __attribute__((ext_vector_type(8)))  float    v8f;

#define NB   8      // batches per block (one wave32 per batch in conv2)
#define TT   50
#define INC  4
#define CH   64
#define DD   128
#define SS   8
#define KTOT 192    // CH * 3 (im2col K)

// ---- dynamic LDS layout (bytes) ----
#define OFF_HIN   0        // NB*52*INC f32            = 6656
#define OFF_W1    6656     // 64*4*3 f32               = 3072
#define OFF_B1    9728     // 64 f32
#define OFF_B2    9984     // 128 f32
#define OFF_LNG   10496    // 128 f32
#define OFF_LNB   11008    // 128 f32
#define OFF_RATE  11520    // NB f32
#define OFF_LI    11552    // NB i32   (pad to 11648)
#define OFF_POOL  11648    // NB*8*129 f32             = 33024 -> ends 44672
#define OFF_H1    44672    // NB*52*64 f16             = 53248 -> ends 97920
#define OFF_W2T   97920    // 128*192 f16              = 49152 -> ends 147072
// phase-B overlay of the dead h1/W2t region:
#define OFF_SPK   44672    // NB*8*128 f16             = 16384 -> ends 61056
#define OFF_PJW   61056    // 128*128 f16              = 32768 -> ends 93824
#define SMEM_BYTES 147072

__device__ __forceinline__ float gelu_exact(float x) {
    return 0.5f * x * (1.0f + erff(x * 0.70710678118654752f));
}

extern "C" __global__ __launch_bounds__(256)
void snn_encoder_fused(const float* __restrict__ hist,   // [B,50,4]
                       const int*   __restrict__ vmask,  // [B,50]
                       const float* __restrict__ w1,     // [64,4,3]
                       const float* __restrict__ b1,     // [64]
                       const float* __restrict__ w2,     // [128,64,3]
                       const float* __restrict__ b2,     // [128]
                       const float* __restrict__ lng,    // [128]
                       const float* __restrict__ lnb,    // [128]
                       const float* __restrict__ pw,     // [128,128]
                       const float* __restrict__ pbv,    // [128]
                       float* __restrict__ out,          // [B*128 | B]
                       int B)
{
    extern __shared__ char smem[];
    float*    s_hin  = (float*)(smem + OFF_HIN);
    float*    s_w1   = (float*)(smem + OFF_W1);
    float*    s_b1   = (float*)(smem + OFF_B1);
    float*    s_b2   = (float*)(smem + OFF_B2);
    float*    s_lng  = (float*)(smem + OFF_LNG);
    float*    s_lnb  = (float*)(smem + OFF_LNB);
    float*    s_rate = (float*)(smem + OFF_RATE);
    int*      s_li   = (int*)  (smem + OFF_LI);
    float*    s_pool = (float*)(smem + OFF_POOL);   // [NB][8][129] padded stride
    _Float16* s_h1   = (_Float16*)(smem + OFF_H1);  // [NB][52][64] rows t=-1..50
    _Float16* s_w2t  = (_Float16*)(smem + OFF_W2T); // [128][192] kk = k*64+c
    _Float16* s_spk  = (_Float16*)(smem + OFF_SPK); // [NB][8][128]
    _Float16* s_pjw  = (_Float16*)(smem + OFF_PJW); // [128][128]

    const int tid     = threadIdx.x;
    const int lane    = tid & 31;
    const int wave    = tid >> 5;
    const int halfsel = lane >> 4;      // 0 for lanes 0-15, 1 for 16-31
    const int m       = lane & 15;
    const int b0      = blockIdx.x * NB;

    // ================= Phase 1: stage inputs / weights ==================
    for (int i = tid; i < NB*TT*INC; i += 256) {
        int b = i / (TT*INC), r = i % (TT*INC);
        int t = r / INC, c = r % INC;
        s_hin[(b*52 + t + 1)*INC + c] = hist[(size_t)(b0+b)*TT*INC + t*INC + c];
    }
    for (int i = tid; i < NB*INC; i += 256) {       // zero t=-1 and t=50 pad rows
        int b = i / INC, c = i % INC;
        s_hin[(b*52 + 0)*INC + c]  = 0.f;
        s_hin[(b*52 + 51)*INC + c] = 0.f;
    }
    for (int i = tid; i < CH*INC*3; i += 256) s_w1[i] = w1[i];
    if (tid < CH) s_b1[tid] = b1[tid];
    if (tid < DD) { s_b2[tid] = b2[tid]; s_lng[tid] = lng[tid]; s_lnb[tid] = lnb[tid]; }
    for (int i = tid; i < NB*SS*129; i += 256) s_pool[i] = 0.f;
    if (tid < NB) {
        s_rate[tid] = 0.f;
        int sum = 0;
        const int* mrow = vmask + (size_t)(b0 + tid)*TT;
        for (int t = 0; t < TT; ++t) sum += mrow[t];
        sum = min(max(sum, 1), SS);
        s_li[tid] = sum - 1;
    }
    // repack conv2_w [d][c][k] -> [d][k*64+c] f16 (32-wide K chunks contiguous)
    for (int i = tid; i < DD*KTOT; i += 256) {
        int d = i / KTOT, kk = i % KTOT;
        int k = kk >> 6, c = kk & 63;
        s_w2t[d*KTOT + kk] = (_Float16)w2[d*KTOT + c*3 + k];
    }
    __syncthreads();

    // ================= Phase 2: conv1 + GELU -> h1 (f16) ================
    for (int i = tid; i < NB*TT*CH; i += 256) {
        int b = i / (TT*CH), r = i % (TT*CH);
        int t = r / CH, c = r % CH;
        float acc = s_b1[c];
        const float* wrow = s_w1 + c*12;                  // [i][k]
        const float* hrow = s_hin + (b*52 + t)*INC;       // t-1+k -> row t+k
        #pragma unroll
        for (int k = 0; k < 3; ++k)
            #pragma unroll
            for (int ii = 0; ii < INC; ++ii)
                acc += hrow[k*INC + ii] * wrow[ii*3 + k];
        s_h1[(b*52 + t + 1)*CH + c] = (_Float16)gelu_exact(acc);
    }
    for (int i = tid; i < NB*CH; i += 256) {              // zero pad rows
        int b = i / CH, c = i % CH;
        s_h1[(b*52 + 0)*CH + c]  = (_Float16)0.f;
        s_h1[(b*52 + 51)*CH + c] = (_Float16)0.f;
    }
    __syncthreads();

    // ====== Phase 3: conv2 via WMMA f16 + GELU + adaptive avg pool ======
    {
        const int b = wave;                               // one batch per wave
        for (int tt = 0; tt < 4; ++tt) {                  // t-tiles (rows t=tt*16+M)
            v16h a[6];
            #pragma unroll
            for (int j = 0; j < 6; ++j) {                 // K chunks of 32
                int jj = j >> 1;                          // conv tap (t-1+jj)
                int cb = (j & 1) * 32;                    // channel sub-block
                int hrow = tt*16 + m + jj;                // (t-1+jj)+1 store index
                hrow = min(hrow, 51);                     // clamp into zero pad
                const _Float16* p = s_h1 + (b*52 + hrow)*CH + cb + halfsel*16;
                v8h lo = *(const v8h*)(p);                // K{0..7} / K{16..23}
                v8h hi = *(const v8h*)(p + 8);            // K{8..15} / K{24..31}
                a[j] = __builtin_shufflevector(lo, hi, 0,1,2,3,4,5,6,7,
                                                        8,9,10,11,12,13,14,15);
            }
            for (int nd = 0; nd < 8; ++nd) {              // d-tiles
                v8f c = {};
                #pragma unroll
                for (int j = 0; j < 6; ++j) {
                    const _Float16* q = s_w2t + (nd*16 + m)*KTOT + j*32 + halfsel*16;
                    v16h bv = *(const v16h*)(q);          // contiguous 16 K per lane
                    c = __builtin_amdgcn_wmma_f32_16x16x32_f16(
                            false, a[j], false, bv, (short)0, c, false, false);
                }
                int d = nd*16 + m;
                float bias = s_b2[d];
                #pragma unroll
                for (int r = 0; r < 8; ++r) {
                    int t = tt*16 + r + halfsel*8;        // C layout: M=r(+8)
                    if (t < TT) {
                        float y = gelu_exact(c[r] + bias) * (1.0f/7.0f);
                        int sb = (t*8)/50;                // primary bin
                        // race-free: lanes differ in d or bin (|Δt|=8 > width 7)
                        s_pool[(b*8 + sb)*129 + d] += y;
                        if (sb < 7 && ((sb+1)*50)/8 <= t) // overlap bin
                            s_pool[(b*8 + sb + 1)*129 + d] += y;
                    }
                }
            }
        }
    }
    __syncthreads();

    // ================= Phase 4: LayerNorm over D ========================
    if (tid < NB*SS) {
        float* row = s_pool + tid*129;
        float mu = 0.f;
        for (int d = 0; d < DD; ++d) mu += row[d];
        mu *= (1.0f/128.0f);
        float var = 0.f;
        for (int d = 0; d < DD; ++d) { float df = row[d]-mu; var += df*df; }
        var *= (1.0f/128.0f);
        float inv = rsqrtf(var + 1e-5f);
        for (int d = 0; d < DD; ++d)
            row[d] = (row[d]-mu)*inv*s_lng[d] + s_lnb[d];
    }
    __syncthreads();

    // ===== Phase 5: two LIF layers (sequential in s) + spike rate =======
    for (int p = tid; p < NB*DD; p += 256) {
        int b = p / DD, d = p % DD;
        float v1 = 0.f, v2 = 0.f, rsum = 0.f;
        #pragma unroll
        for (int s = 0; s < SS; ++s) {
            float x  = s_pool[(b*8 + s)*129 + d];
            v1 = 0.5f*(v1 + x);                         // v += (x-v)/tau, tau=2
            float s1 = (v1 >= 1.0f) ? 1.0f : 0.0f;      // heaviside(v - v_th)
            v1 *= (1.0f - s1);                          // hard reset
            v2 = 0.5f*(v2 + s1);
            float s2 = (v2 >= 1.0f) ? 1.0f : 0.0f;
            v2 *= (1.0f - s2);
            s_spk[(b*8 + s)*DD + d] = (_Float16)s2;     // exact 0/1 in f16
            rsum += s2;
        }
        atomicAdd(&s_rate[b], rsum);                    // small-int sums: exact
    }
    for (int i = tid; i < DD*DD; i += 256) s_pjw[i] = (_Float16)pw[i];
    __syncthreads();

    // ========== Phase 6: spike_rate out + out_proj (WMMA) gather ========
    if (tid < NB) out[(size_t)B*DD + b0 + tid] = s_rate[tid] * (1.0f/1024.0f);
    {
        const int n = wave*16 + m;                       // one d-out tile per wave
        v8f c = {};
        #pragma unroll
        for (int j = 0; j < 4; ++j) {                    // K = 128 -> 4 chunks
            int bb = m & 7;                              // rows 8..15 duplicate 0..7
            const _Float16* pa = s_spk + (bb*8 + s_li[bb])*DD + j*32 + halfsel*16;
            v8h lo = *(const v8h*)(pa);
            v8h hi = *(const v8h*)(pa + 8);
            v16h av = __builtin_shufflevector(lo, hi, 0,1,2,3,4,5,6,7,
                                                       8,9,10,11,12,13,14,15);
            v16h bv = *(const v16h*)(s_pjw + n*DD + j*32 + halfsel*16);
            c = __builtin_amdgcn_wmma_f32_16x16x32_f16(
                    false, av, false, bv, (short)0, c, false, false);
        }
        float bias = pbv[n];
        if (halfsel == 0) {                              // M = r in [0,8): valid b
            #pragma unroll
            for (int r = 0; r < 8; ++r)
                out[(size_t)(b0 + r)*DD + n] = c[r] + bias;
        }
    }
}

extern "C" void kernel_launch(void* const* d_in, const int* in_sizes, int n_in,
                              void* d_out, int out_size, void* d_ws, size_t ws_size,
                              hipStream_t stream) {
    (void)n_in; (void)out_size; (void)d_ws; (void)ws_size;
    const float* hist = (const float*)d_in[0];
    const int*   msk  = (const int*)  d_in[1];
    const float* w1   = (const float*)d_in[2];
    const float* b1   = (const float*)d_in[3];
    const float* w2   = (const float*)d_in[4];
    const float* b2   = (const float*)d_in[5];
    const float* lng  = (const float*)d_in[6];
    const float* lnb  = (const float*)d_in[7];
    const float* pw   = (const float*)d_in[8];
    const float* pb   = (const float*)d_in[9];
    int B = in_sizes[0] / (TT*INC);                      // 16384
    (void)hipFuncSetAttribute((const void*)snn_encoder_fused,
                              hipFuncAttributeMaxDynamicSharedMemorySize,
                              SMEM_BYTES);
    snn_encoder_fused<<<B / NB, 256, SMEM_BYTES, stream>>>(
        hist, msk, w1, b1, w2, b2, lng, lnb, pw, pb, (float*)d_out, B);
}
